// BottomLevelDecoderRNN_90134183674508
// MI455X (gfx1250) — matrix operand order repaired and checked by reading, not compiled, not run on hardware
//
#include <hip/hip_runtime.h>
#include <math.h>

// ---------------- problem constants (from reference) ----------------
constexpr int B    = 256;
constexpr int BARS = 16;
constexpr int CH   = 512;
constexpr int H    = 1024;   // hidden
constexpr int V    = 130;
constexpr int T    = 256;
constexpr int G3   = 3 * H;  // packed gates i,g,o (f gate is dead in _lstm_h)
constexpr long BH  = (long)B * H;

typedef __bf16 bf16;
typedef __attribute__((ext_vector_type(16))) __bf16 v16bf;
typedef __attribute__((ext_vector_type(8)))  float  v8f;

#define BM 128
#define BN 128
#define BK 64

__device__ __forceinline__ float sigmoidf_(float x) { return 1.0f / (1.0f + __expf(-x)); }

union Frag { v16bf v; uint4 u[2]; };

// ---- CDNA5 async global->LDS copy (ASYNCcnt path), with sync fallback ----
// Builtin signature (from hipcc diagnostic): param0 = int4 in device (AS1),
// param1 = LDS (AS3), then imm offset, imm cpol.
#if __has_builtin(__builtin_amdgcn_global_load_async_to_lds_b128)
#define HAVE_ASYNC_LDS 1
typedef int v4i_ __attribute__((vector_size(4 * sizeof(int))));
typedef __attribute__((address_space(1))) v4i_* as1_v4i;
typedef __attribute__((address_space(3))) v4i_* as3_v4i;
__device__ __forceinline__ void async_cp16(const bf16* g, bf16* l) {
    __builtin_amdgcn_global_load_async_to_lds_b128(
        (as1_v4i)(void*)g, (as3_v4i)(void*)l, 0, 0);
}
#else
#define HAVE_ASYNC_LDS 0
#endif

#if __has_builtin(__builtin_amdgcn_s_wait_asynccnt)
#define WAIT_ASYNC() __builtin_amdgcn_s_wait_asynccnt(0)
#else
#define WAIT_ASYNC() asm volatile("s_wait_asynccnt 0" ::: "memory")
#endif

// ---------------------------------------------------------------
// Generic bf16 WMMA GEMM:  C[M=256 x N] = A[256 x K] * W[N x K]^T
// block tile 128x128, K-chunk 64, 8 waves (wave32) as 2(M) x 4(N),
// wave tile 64x32 -> 4x2 tiles of v_wmma_f32_16x16x32_bf16.
// Double-buffered LDS, staged with global_load_async_to_lds_b128.
// mode 0: C = acc (f32, ldc = N)          (gates scratch)
// mode 1: C = acc + bias[col], col < N    (output projection, strided C)
// ---------------------------------------------------------------
__global__ __launch_bounds__(256) void wmma_gemm(
    const bf16* __restrict__ A, long lda,
    const bf16* __restrict__ W, long ldb,
    int N, int K,
    float* __restrict__ C, long ldc,
    const float* __restrict__ bias, int mode)
{
    __shared__ __align__(16) bf16 As[2][BM][BK];
    __shared__ __align__(16) bf16 Bs[2][BN][BK];

    const int tid  = threadIdx.x;
    const int lane = tid & 31;
    const int wave = tid >> 5;
    const int wm   = wave >> 2;   // 0..1  (M dir)
    const int wn   = wave & 3;    // 0..3  (N dir)
    const long blockN = (long)blockIdx.x * BN;
    const long blockM = (long)blockIdx.y * BM;

    v8f acc[4][2] = {};

    // Stage one 128x64 tile of A and of W into LDS buffer `buf`.
    // 1024 16B chunks each; 256 threads x 4 chunks.
    auto stage = [&](int buf, int k0) {
#pragma unroll
        for (int p = 0; p < 4; ++p) {
            int chunk = tid + p * 256;
            int r  = chunk >> 3;
            int cc = (chunk & 7) * 8;
            const bf16* ga = A + (blockM + r) * lda + k0 + cc;
            long gn = blockN + r; if (gn >= N) gn = N - 1;   // clamp (stores masked later)
            const bf16* gb = W + gn * ldb + k0 + cc;
#if HAVE_ASYNC_LDS
            async_cp16(ga, &As[buf][r][cc]);
            async_cp16(gb, &Bs[buf][r][cc]);
#else
            *(uint4*)&As[buf][r][cc] = *(const uint4*)ga;
            *(uint4*)&Bs[buf][r][cc] = *(const uint4*)gb;
#endif
        }
    };

    const int nIter = K / BK;
    stage(0, 0);
    for (int it = 0; it < nIter; ++it) {
        const int buf = it & 1;
        WAIT_ASYNC();        // this wave's async copies (tile `it`) have landed in LDS
        __syncthreads();     // => every wave's copies landed; every wave done with tile it-1
        if (it + 1 < nIter)  // overlap next tile's L2 stream with the WMMAs below
            stage(buf ^ 1, (it + 1) * BK);

#pragma unroll
        for (int kk = 0; kk < BK; kk += 32) {
            Frag a[4], b[2];
            // A frag (ISA 16-bit A 16x32): lanes 0-15 rows hold K 0..7 & 16..23,
            // lanes 16-31 hold K 8..15 & 24..31.
            const int ak = kk + ((lane >> 4) << 3);
#pragma unroll
            for (int mt = 0; mt < 4; ++mt) {
                int ar = wm * 64 + mt * 16 + (lane & 15);
                a[mt].u[0] = *(const uint4*)&As[buf][ar][ak];
                a[mt].u[1] = *(const uint4*)&As[buf][ar][ak + 16];
            }
            // B frag: lanes 0-15 = cols, K 0..7 & 16..23; lanes 16-31 K 8..15 & 24..31.
            const int bk = kk + ((lane >> 4) << 3);
#pragma unroll
            for (int nt = 0; nt < 2; ++nt) {
                int br = wn * 32 + nt * 16 + (lane & 15);
                b[nt].u[0] = *(const uint4*)&Bs[buf][br][bk];
                b[nt].u[1] = *(const uint4*)&Bs[buf][br][bk + 16];
            }
#pragma unroll
            for (int mt = 0; mt < 4; ++mt)
#pragma unroll
                for (int nt = 0; nt < 2; ++nt)
                    acc[mt][nt] = __builtin_amdgcn_wmma_f32_16x16x32_bf16(
                        false, a[mt].v, false, b[nt].v,
                        (short)0, acc[mt][nt], false, false);
        }
    }

    // Epilogue: C/D layout -> VGPR e: lanes 0-15 M=e, lanes 16-31 M=8+e; col = lane&15.
#pragma unroll
    for (int mt = 0; mt < 4; ++mt) {
        long rbase = blockM + wm * 64 + mt * 16 + ((lane >> 4) << 3);
#pragma unroll
        for (int nt = 0; nt < 2; ++nt) {
            long col = blockN + wn * 32 + nt * 16 + (lane & 15);
#pragma unroll
            for (int e = 0; e < 8; ++e) {
                long r = rbase + e;
                if (mode == 1) {
                    if (col < N) C[r * ldc + col] = acc[mt][nt][e] + bias[col];
                } else {
                    C[r * ldc + col] = acc[mt][nt][e];
                }
            }
        }
    }
}

// ---------------- weight / input prep kernels ----------------

// f32 -> bf16 flat copy
__global__ void f32_to_bf16_k(bf16* dst, const float* src, long n) {
    long i = (long)blockIdx.x * 256 + threadIdx.x;
    if (i < n) dst[i] = (bf16)src[i];
}

// Pack [Wih | Whh] rows for gates i,g,o only (skip dead f gate).
// dst is [3H x (Kih+Khh)] bf16; packed row r maps to source gate row r + (r>=H ? H : 0).
__global__ void pack_gates_k(bf16* dst, const float* Wih, int Kih,
                             const float* Whh, int Khh, long total) {
    long idx = (long)blockIdx.x * 256 + threadIdx.x;
    if (idx >= total) return;
    int Kt = Kih + Khh;
    long r = idx / Kt;
    int  k = (int)(idx % Kt);
    long sr = r + (r >= H ? H : 0);
    float v = (k < Kih) ? Wih[sr * Kih + k] : Whh[sr * Khh + (k - Kih)];
    dst[idx] = (bf16)v;
}

// ---------------- per-step elementwise kernels ----------------

// X1[i] = [emb[i,0,:] (bcast), c_ptr, h1[i]]  -> (3,B,2048)
__global__ void concat_x1_k(bf16* X1, const bf16* embb, const bf16* cbf,
                            const bf16* h1b, int bar) {
    int i = blockIdx.z;
    long idx = (long)blockIdx.x * 256 + threadIdx.x;
    if (idx >= (long)B * 2048) return;
    int b = (int)(idx >> 11), col = (int)(idx & 2047);
    bf16 v;
    if (col < 512)        v = embb[((long)i * V + 0) * CH + col];
    else if (col < 1024)  v = cbf[((long)b * BARS + bar) * CH + (col - 512)];
    else                  v = h1b[((long)i * B + b) * H + (col - 1024)];
    X1[(long)i * B * 2048 + idx] = v;
}

// Xctx = [u0,u1,u2,ctx]  -> (B,4096)
__global__ void concat_xctx_k(bf16* X, const bf16* ub, const bf16* ctxb) {
    long idx = (long)blockIdx.x * 256 + threadIdx.x;
    if (idx >= (long)B * 4096) return;
    int b = (int)(idx >> 12), col = (int)(idx & 4095);
    bf16 v;
    if (col < 3072) {
        int seg = col >> 10, j = col & 1023;
        v = ub[((long)seg * B + b) * H + j];
    } else v = ctxb[(long)b * H + (col - 3072)];
    X[idx] = v;
}

// Xh2 = [ctx, c_ptr, h2[i]] -> (B,2560)
__global__ void concat_xh2_k(bf16* X, const bf16* ctxb, const bf16* cbf,
                             const bf16* h2b, int bar, int i) {
    long idx = (long)blockIdx.x * 256 + threadIdx.x;
    if (idx >= (long)B * 2560) return;
    int b = (int)(idx / 2560), col = (int)(idx % 2560);
    bf16 v;
    if (col < 1024)       v = ctxb[(long)b * H + col];
    else if (col < 1536)  v = cbf[((long)b * BARS + bar) * CH + (col - 1024)];
    else                  v = h2b[((long)i * B + b) * H + (col - 1536)];
    X[idx] = v;
}

// Xu = [emb[i][tgt], c_ptr, u[i]] -> (B,2048)
__global__ void concat_xu_k(bf16* X, const bf16* embb, const bf16* cbf,
                            const bf16* ub, const int* target, int bar, int i, int t) {
    long idx = (long)blockIdx.x * 256 + threadIdx.x;
    if (idx >= (long)B * 2048) return;
    int b = (int)(idx >> 11), col = (int)(idx & 2047);
    bf16 v;
    if (col < 512) {
        int tg = target[((long)i * B + b) * T + t];
        v = embb[((long)i * V + tg) * CH + col];
    } else if (col < 1024) v = cbf[((long)b * BARS + bar) * CH + (col - 512)];
    else                   v = ub[((long)i * B + b) * H + (col - 1024)];
    X[idx] = v;
}

// h = sigmoid(o+bo) * tanh(sigmoid(i+bi) * tanh(g+bg)); packed gate cols [i|g|o].
// Optional outputs: hf (f32), hb/hb2 (bf16), s = u + h (bf16).
__global__ void act_lstm_k(const float* G, long gStride, const float* bias, long bStride,
                           float* hf, bf16* hb, bf16* hb2,
                           const float* uadd, bf16* sb) {
    int z = blockIdx.z;
    long idx = (long)blockIdx.x * 256 + threadIdx.x;
    if (idx >= BH) return;
    int b = (int)(idx >> 10), j = (int)(idx & 1023);
    const float* g  = G + (long)z * gStride;
    const float* bb = bias + (long)z * bStride;
    float gi = g[(long)b * G3 + j]        + bb[j];            // i gate
    float gg = g[(long)b * G3 + H + j]    + bb[2 * H + j];    // g gate
    float go = g[(long)b * G3 + 2 * H + j]+ bb[3 * H + j];    // o gate
    float h = sigmoidf_(go) * tanhf(sigmoidf_(gi) * tanhf(gg));
    long o = (long)z * BH + idx;
    if (hf)  hf[o] = h;
    hb[o] = (bf16)h;
    if (hb2) hb2[o] = (bf16)h;
    if (sb)  sb[idx] = (bf16)(uadd[idx] + h);
}

// reset state: hidden = tanh(G + hid_b) broadcast into h1[0..2], h2[0..2], ctx (bf16)
__global__ void hidden_act_k(const float* G, const float* hid_b,
                             bf16* h1b, bf16* h2b, bf16* ctxb) {
    long idx = (long)blockIdx.x * 256 + threadIdx.x;
    if (idx >= BH) return;
    int j = (int)(idx & 1023);
    bf16 v = (bf16)tanhf(G[idx] + hid_b[j]);
    h1b[idx] = v; h1b[BH + idx] = v; h1b[2 * BH + idx] = v;
    h2b[idx] = v; h2b[BH + idx] = v; h2b[2 * BH + idx] = v;
    ctxb[idx] = v;
}

// ---------------- host driver ----------------
extern "C" void kernel_launch(void* const* d_in, const int* in_sizes, int n_in,
                              void* d_out, int out_size, void* d_ws, size_t ws_size,
                              hipStream_t stream) {
    (void)in_sizes; (void)n_in; (void)out_size; (void)ws_size;
    const float* c_in   = (const float*)d_in[0];
    const float* l1Wih  = (const float*)d_in[1];
    const float* l1Whh  = (const float*)d_in[2];
    const float* l1b    = (const float*)d_in[3];
    const float* cWih   = (const float*)d_in[4];
    const float* cWhh   = (const float*)d_in[5];
    const float* cb     = (const float*)d_in[6];
    const float* l2Wih  = (const float*)d_in[7];
    const float* l2Whh  = (const float*)d_in[8];
    const float* l2b    = (const float*)d_in[9];
    const float* outW   = (const float*)d_in[10];
    const float* outb   = (const float*)d_in[11];
    const float* hidW   = (const float*)d_in[12];
    const float* hidb   = (const float*)d_in[13];
    const float* emb    = (const float*)d_in[14];
    const int*   target = (const int*)d_in[15];
    float* out = (float*)d_out;

    char* ws = (char*)d_ws;
    size_t off = 0;
    auto alloc = [&](size_t bytes) -> char* {
        char* p = ws + off;
        off = (off + bytes + 255) & ~(size_t)255;
        return p;
    };
    // bf16 weights (~150 MB -> L2 resident), activations, scratch
    bf16* cbf   = (bf16*)alloc(2L * B * BARS * CH);
    bf16* embb  = (bf16*)alloc(2L * 3 * V * CH);
    bf16* hidWp = (bf16*)alloc(2L * H * CH);
    bf16* outWb = (bf16*)alloc(2L * 3 * V * H);
    bf16* W1p   = (bf16*)alloc(2L * 3 * G3 * 2048);
    bf16* Wctxp = (bf16*)alloc(2L * G3 * 4096);
    bf16* W2p   = (bf16*)alloc(2L * 3 * G3 * 2560);
    bf16* h1b   = (bf16*)alloc(2L * 3 * BH);
    bf16* h2b   = (bf16*)alloc(2L * 3 * BH);
    bf16* ub    = (bf16*)alloc(2L * 3 * BH);
    float* uf   = (float*)alloc(4L * 3 * BH);
    bf16* ctxb  = (bf16*)alloc(2L * BH);
    bf16* sb    = (bf16*)alloc(2L * BH);
    bf16* X1    = (bf16*)alloc(2L * 3 * B * 2048);
    bf16* Xctx  = (bf16*)alloc(2L * B * 4096);
    bf16* Xh2   = (bf16*)alloc(2L * B * 2560);
    bf16* Xu    = (bf16*)alloc(2L * B * 2048);
    float* G1   = (float*)alloc(4L * 3 * B * G3);
    float* Gctx = (float*)alloc(4L * B * G3);
    float* Gh2  = (float*)alloc(4L * B * G3);
    float* Gu   = (float*)alloc(4L * B * G3);
    float* Ghid = (float*)alloc(4L * BH);

    auto cdiv = [](long a, long b) { return (int)((a + b - 1) / b); };
    auto gemm = [&](const bf16* A, long lda, const bf16* W, long ldb, int N, int K,
                    float* C, long ldc, const float* bias, int mode) {
        dim3 grid(cdiv(N, BN), 256 / BM);
        wmma_gemm<<<grid, 256, 0, stream>>>(A, lda, W, ldb, N, K, C, ldc, bias, mode);
    };

    // ---- prep: f32 -> bf16, pack [Wih|Whh] with f-gate rows dropped ----
    f32_to_bf16_k<<<cdiv((long)B * BARS * CH, 256), 256, 0, stream>>>(cbf, c_in, (long)B * BARS * CH);
    f32_to_bf16_k<<<cdiv(3L * V * CH, 256), 256, 0, stream>>>(embb, emb, 3L * V * CH);
    f32_to_bf16_k<<<cdiv((long)H * CH, 256), 256, 0, stream>>>(hidWp, hidW, (long)H * CH);
    f32_to_bf16_k<<<cdiv(3L * V * H, 256), 256, 0, stream>>>(outWb, outW, 3L * V * H);
    for (int i = 0; i < 3; ++i)
        pack_gates_k<<<cdiv((long)G3 * 2048, 256), 256, 0, stream>>>(
            W1p + (long)i * G3 * 2048, l1Wih + (long)i * 4 * H * 1024, 1024,
            l1Whh + (long)i * 4 * H * 1024, 1024, (long)G3 * 2048);
    pack_gates_k<<<cdiv((long)G3 * 4096, 256), 256, 0, stream>>>(
        Wctxp, cWih, 3072, cWhh, 1024, (long)G3 * 4096);
    for (int i = 0; i < 3; ++i)
        pack_gates_k<<<cdiv((long)G3 * 2560, 256), 256, 0, stream>>>(
            W2p + (long)i * G3 * 2560, l2Wih + (long)i * 4 * H * 1536, 1536,
            l2Whh + (long)i * 4 * H * 1024, 1024, (long)G3 * 2560);

    const int actBlocks = cdiv(BH, 256);

    // ---- sequential recurrence over T steps ----
    for (int t = 0; t < T; ++t) {
        int bar = t >> 4;
        if ((t & 15) == 0) {
            int prev = bar > 0 ? bar - 1 : 0;
            // hidden = tanh(c[:,prev] @ hidW^T + hidb): M=256, N=1024, K=512
            gemm(cbf + (long)prev * CH, (long)BARS * CH, hidWp, CH, H, CH, Ghid, H, nullptr, 0);
            hidden_act_k<<<actBlocks, 256, 0, stream>>>(Ghid, hidb, h1b, h2b, ctxb);
        }
        // h1[i] = lstm([notes0_i, c_ptr], h1[i])  (3 independent GEMMs)
        concat_x1_k<<<dim3(cdiv((long)B * 2048, 256), 1, 3), 256, 0, stream>>>(X1, embb, cbf, h1b, bar);
        for (int i = 0; i < 3; ++i)
            gemm(X1 + (long)i * B * 2048, 2048, W1p + (long)i * G3 * 2048, 2048,
                 G3, 2048, G1 + (long)i * B * G3, G3, nullptr, 0);
        // -> h1 (bf16 carry) and u = h1 (f32 + bf16)
        act_lstm_k<<<dim3(actBlocks, 1, 3), 256, 0, stream>>>(
            G1, (long)B * G3, l1b, 4L * H, uf, h1b, ub, nullptr, nullptr);

        for (int i = 0; i < 3; ++i) {
            // ctx = lstm([u0,u1,u2], ctx)
            concat_xctx_k<<<cdiv((long)B * 4096, 256), 256, 0, stream>>>(Xctx, ub, ctxb);
            gemm(Xctx, 4096, Wctxp, 4096, G3, 4096, Gctx, G3, nullptr, 0);
            act_lstm_k<<<dim3(actBlocks, 1, 1), 256, 0, stream>>>(
                Gctx, 0, cb, 0, nullptr, ctxb, nullptr, nullptr, nullptr);
            // h2[i] = lstm([ctx, c_ptr], h2[i])
            concat_xh2_k<<<cdiv((long)B * 2560, 256), 256, 0, stream>>>(Xh2, ctxb, cbf, h2b, bar, i);
            gemm(Xh2, 2560, W2p + (long)i * G3 * 2560, 2560, G3, 2560, Gh2, G3, nullptr, 0);
            // u[i] = lstm([emb[i][tgt], c_ptr], u[i])  (weights = lstm1 cell 1)
            concat_xu_k<<<cdiv((long)B * 2048, 256), 256, 0, stream>>>(Xu, embb, cbf, ub, target, bar, i, t);
            gemm(Xu, 2048, W1p + 1L * G3 * 2048, 2048, G3, 2048, Gu, G3, nullptr, 0);
            // h2 activation, and s = u[i] (pre-update) + h2[i]
            act_lstm_k<<<dim3(actBlocks, 1, 1), 256, 0, stream>>>(
                Gh2, 0, l2b + (long)i * 4 * H, 0, nullptr, h2b + (long)i * BH, nullptr,
                uf + (long)i * BH, sb);
            // u[i] activation (overwrites u after s was formed)
            act_lstm_k<<<dim3(actBlocks, 1, 1), 256, 0, stream>>>(
                Gu, 0, l1b + 4L * H, 0, uf + (long)i * BH, ub + (long)i * BH, nullptr,
                nullptr, nullptr);
            // outs[i] = s @ outW[i]^T + outb[i]  -> d_out[i, :, t, :]
            gemm(sb, H, outWb + (long)i * V * H, H, V, H,
                 out + (long)i * B * T * V + (long)t * V, (long)T * V, outb + (long)i * V, 1);
        }
    }
}
